// MambaWeatherBlock_10531259810087
// MI455X (gfx1250) — matrix operand order repaired and checked by reading, not compiled, not run on hardware
//
#include <hip/hip_runtime.h>

// ---------------- problem constants ----------------
#define B_   4
#define L_   2048
#define D_   768
#define DI_  1536
#define NST  16          // d_state
#define KC   4           // conv width
#define RR   48          // dt_rank
#define RRP  64          // dt_rank padded to K multiple of 32
#define WW   64          // weather dim
#define BL   (B_ * L_)   // 8192 rows (multiple of 32)

typedef __attribute__((ext_vector_type(16))) __bf16 v16bf;
typedef __attribute__((ext_vector_type(8)))  float  v8f;
typedef __attribute__((ext_vector_type(4)))  unsigned int v4u;

union FragCast { v16bf v; v4u u[2]; };

// ---------------- unguarded fragment loaders (bf16, 16x16x32 WMMA) ----------------
// A: 16x32 (MxK), lane = M mod 16; per lane two contiguous 16B runs at +0 and +32B.
// p must already point at rowBase + kbase + laneHi*8.
__device__ inline v16bf load_fragA(const __bf16* __restrict__ p) {
  FragCast fc;
  fc.u[0] = *(const v4u*)(p);        // K = k0 .. k0+7
  fc.u[1] = *(const v4u*)(p + 16);   // K = k0+16 .. k0+23
  return fc.v;
}
// B: 32x16 (KxN) from row-major W[N][K]; lane = N mod 16; two contiguous 16B runs at +0 and +32B.
// p points at rowBase + kbase + laneHi*8.
__device__ inline v16bf load_fragB(const __bf16* __restrict__ p) {
  FragCast fc;
  fc.u[0] = *(const v4u*)(p);
  fc.u[1] = *(const v4u*)(p + 16);
  return fc.v;
}

__device__ inline void store_tile(const v8f& acc, float* __restrict__ C, int r0, int c,
                                  int Nn, const float* __restrict__ bias, int act, int laneHi) {
  if (c < Nn) {
    const float bv = bias ? bias[c] : 0.0f;
#pragma unroll
    for (int v = 0; v < 8; ++v) {
      const int r = r0 + v + (laneHi << 3);
      float val = acc[v] + bv;
      if (act == 1) val = (val > 20.0f) ? val : log1pf(expf(val));
      C[(size_t)r * Nn + c] = val;
    }
  }
}

// ---------------- bf16 WMMA GEMM: C[M,Nn] = A[M,Kpad] * W[Npad,Kpad]^T ----------------
// Requirements (guaranteed by caller): M % 32 == 0, Kpad % 32 == 0, Npad % 64 == 0,
// A and W zero-padded to those shapes. Wave computes 32(M) x 64(N): 8 WMMAs / K-step.
__global__ void __launch_bounds__(256)
wmma_gemm_bf16(const __bf16* __restrict__ A, const __bf16* __restrict__ Wm,
               float* __restrict__ C, const float* __restrict__ bias,
               int M, int Nn, int Npad, int Kpad, int act) {
  const int lane   = threadIdx.x & 31;
  const int laneLo = lane & 15;
  const int laneHi = lane >> 4;
  const int mtiles = M >> 5;            // 32-row tiles
  const int n64    = Npad >> 6;         // 64-col tiles
  const int total  = mtiles * n64;
  const int nwaves = gridDim.x * (blockDim.x >> 5);
  int wave = blockIdx.x * (blockDim.x >> 5) + (threadIdx.x >> 5);

  for (int t = wave; t < total; t += nwaves) {
    const int tm = t / n64;
    const int tn = t - tm * n64;
    const int cn = tn * 64 + laneLo;    // this lane's N column within B rows

    const __bf16* pa0 = A  + (size_t)(tm * 32 + laneLo) * Kpad + (laneHi << 3);
    const __bf16* pa1 = pa0 + (size_t)16 * Kpad;
    const __bf16* pb0 = Wm + (size_t)(cn +  0) * Kpad + (laneHi << 3);
    const __bf16* pb1 = Wm + (size_t)(cn + 16) * Kpad + (laneHi << 3);
    const __bf16* pb2 = Wm + (size_t)(cn + 32) * Kpad + (laneHi << 3);
    const __bf16* pb3 = Wm + (size_t)(cn + 48) * Kpad + (laneHi << 3);

    v8f a00 = (v8f){0.f,0.f,0.f,0.f,0.f,0.f,0.f,0.f};
    v8f a01 = a00, a02 = a00, a03 = a00;
    v8f a10 = a00, a11 = a00, a12 = a00, a13 = a00;

    const int ksteps = Kpad >> 5;
    for (int ks = 0; ks < ksteps; ++ks) {
      v16bf fa0 = load_fragA(pa0);
      v16bf fa1 = load_fragA(pa1);
      v16bf fb0 = load_fragB(pb0);
      v16bf fb1 = load_fragB(pb1);
      v16bf fb2 = load_fragB(pb2);
      v16bf fb3 = load_fragB(pb3);
      a00 = __builtin_amdgcn_wmma_f32_16x16x32_bf16(false, fa0, false, fb0, (short)0, a00, false, false);
      a10 = __builtin_amdgcn_wmma_f32_16x16x32_bf16(false, fa1, false, fb0, (short)0, a10, false, false);
      a01 = __builtin_amdgcn_wmma_f32_16x16x32_bf16(false, fa0, false, fb1, (short)0, a01, false, false);
      a11 = __builtin_amdgcn_wmma_f32_16x16x32_bf16(false, fa1, false, fb1, (short)0, a11, false, false);
      a02 = __builtin_amdgcn_wmma_f32_16x16x32_bf16(false, fa0, false, fb2, (short)0, a02, false, false);
      a12 = __builtin_amdgcn_wmma_f32_16x16x32_bf16(false, fa1, false, fb2, (short)0, a12, false, false);
      a03 = __builtin_amdgcn_wmma_f32_16x16x32_bf16(false, fa0, false, fb3, (short)0, a03, false, false);
      a13 = __builtin_amdgcn_wmma_f32_16x16x32_bf16(false, fa1, false, fb3, (short)0, a13, false, false);
      pa0 += 32; pa1 += 32; pb0 += 32; pb1 += 32; pb2 += 32; pb3 += 32;
    }

    // C/D layout: lane&15 = N column, VGPR v holds M = v + 8*laneHi
    const int r0 = tm * 32;
    const int c0 = tn * 64 + laneLo;
    store_tile(a00, C, r0,      c0,      Nn, bias, act, laneHi);
    store_tile(a01, C, r0,      c0 + 16, Nn, bias, act, laneHi);
    store_tile(a02, C, r0,      c0 + 32, Nn, bias, act, laneHi);
    store_tile(a03, C, r0,      c0 + 48, Nn, bias, act, laneHi);
    store_tile(a10, C, r0 + 16, c0,      Nn, bias, act, laneHi);
    store_tile(a11, C, r0 + 16, c0 + 16, Nn, bias, act, laneHi);
    store_tile(a12, C, r0 + 16, c0 + 32, Nn, bias, act, laneHi);
    store_tile(a13, C, r0 + 16, c0 + 48, Nn, bias, act, laneHi);
  }
}

// ---------------- elementwise helpers ----------------
// generic f32 -> bf16 with zero padding:
// dst is (rowsPad x colsPad) row-major; src element (r,c) at src[r*srcLd + srcOff + c].
__global__ void pack_pad_bf16(const float* __restrict__ src, __bf16* __restrict__ dst,
                              int rows, int cols, int srcLd, int srcOff,
                              int colsPad, int total) {
  for (int i = blockIdx.x * blockDim.x + threadIdx.x; i < total; i += gridDim.x * blockDim.x) {
    const int c = i % colsPad;
    const int r = i / colsPad;
    dst[i] = (r < rows && c < cols) ? (__bf16)src[(size_t)r * srcLd + srcOff + c]
                                    : (__bf16)0.0f;
  }
}

// causal depthwise conv1d + bias + SiLU.  xz: (BL, 3072) rows; x is cols [0,1536)
__global__ void conv_silu_kernel(const float* __restrict__ xz, const float* __restrict__ cw,
                                 const float* __restrict__ cb, float* __restrict__ xconv) {
  const int total = BL * DI_;
  for (int i = blockIdx.x * blockDim.x + threadIdx.x; i < total; i += gridDim.x * blockDim.x) {
    const int d  = i % DI_;
    const int bl = i / DI_;
    const int l  = bl % L_;
    const int b  = bl / L_;
    float acc = cb[d];
#pragma unroll
    for (int j = 0; j < KC; ++j) {
      const int ls = l - (KC - 1) + j;
      if (ls >= 0)
        acc += cw[d * KC + j] * xz[((size_t)(b * L_ + ls)) * (2 * DI_) + d];
    }
    xconv[i] = acc / (1.0f + expf(-acc));   // SiLU
  }
}

// concat [x_conv(1536) | wp(64)] -> bf16 (BL, 1600)
__global__ void concat_pack_kernel(const float* __restrict__ xconv, const float* __restrict__ wp,
                                   __bf16* __restrict__ out) {
  const int total = BL * (DI_ + WW);
  for (int i = blockIdx.x * blockDim.x + threadIdx.x; i < total; i += gridDim.x * blockDim.x) {
    const int c   = i % (DI_ + WW);
    const int row = i / (DI_ + WW);
    out[i] = (c < DI_) ? (__bf16)xconv[(size_t)row * DI_ + c]
                       : (__bf16)wp[(size_t)row * WW + (c - DI_)];
  }
}

// ---------------- fused selective scan + D-skip + z-gate ----------------
// lane = (channel, n); 16-lane shuffle reduction for y = sum_n h*C.
// Software-pipelined loads (next step pre-issued) + global_prefetch_b8 at distance 8.
// Lane n==0 applies  yg = (y + x*D[d]) * silu(z)  and stores bf16 ygate directly.
__global__ void __launch_bounds__(256)
scan_gate_kernel(const float* __restrict__ delta, const float* __restrict__ xconv,
                 const float* __restrict__ xdbl, const float* __restrict__ xz,
                 const float* __restrict__ A_log, const float* __restrict__ Dp,
                 __bf16* __restrict__ ygate) {
  const int gid = blockIdx.x * blockDim.x + threadIdx.x;
  const int n   = gid & (NST - 1);
  const int ch  = gid >> 4;
  if (ch >= B_ * DI_) return;                 // grid sized exactly: uniform
  const int d = ch % DI_;
  const int b = ch / DI_;
  const float Adn = -expf(A_log[d * NST + n]);
  const float Dd  = Dp[d];
  float h = 0.0f;
  const size_t rowbase = (size_t)b * L_;

  // prologue: load step 0
  size_t rl = rowbase;
  float dt = delta[rl * DI_ + d];
  float x  = xconv[rl * DI_ + d];
  float Bv = xdbl[rl * (RR + 2 * NST) + RR + n];
  float Cv = xdbl[rl * (RR + 2 * NST) + RR + NST + n];
  float zv = xz[rl * (2 * DI_) + DI_ + d];

  for (int l = 0; l < L_; ++l) {
    // prefetch 8 rows ahead (pulls the two per-channel streams into cache)
    const size_t rp = rowbase + ((l + 8 < L_) ? (l + 8) : (L_ - 1));
    __builtin_prefetch(&delta[rp * DI_ + d], 0, 3);
    __builtin_prefetch(&xconv[rp * DI_ + d], 0, 3);

    // pre-issue next step's loads (decouples load latency from the recurrence chain)
    const size_t rl1 = rowbase + ((l + 1 < L_) ? (l + 1) : l);
    const float dt_n = delta[rl1 * DI_ + d];
    const float x_n  = xconv[rl1 * DI_ + d];
    const float Bv_n = xdbl[rl1 * (RR + 2 * NST) + RR + n];
    const float Cv_n = xdbl[rl1 * (RR + 2 * NST) + RR + NST + n];
    const float zv_n = xz[rl1 * (2 * DI_) + DI_ + d];

    // recurrence for current step
    h = __expf(dt * Adn) * h + (dt * x) * Bv;
    float y = h * Cv;
    y += __shfl_xor(y, 8);
    y += __shfl_xor(y, 4);
    y += __shfl_xor(y, 2);
    y += __shfl_xor(y, 1);
    if (n == 0) {
      const float g = zv / (1.0f + expf(-zv));          // silu(z)
      ygate[(rowbase + l) * DI_ + d] = (__bf16)((y + x * Dd) * g);
    }

    dt = dt_n; x = x_n; Bv = Bv_n; Cv = Cv_n; zv = zv_n;
  }
}

// ---------------- host launch ----------------
static inline int eblocks(int n) { return (n + 255) / 256; }
static inline int gblocks(int M, int Npad) {
  int t = (M / 32) * (Npad / 64);
  return (t + 7) / 8;   // 8 waves per 256-thread block
}

extern "C" void kernel_launch(void* const* d_in, const int* in_sizes, int n_in,
                              void* d_out, int out_size, void* d_ws, size_t ws_size,
                              hipStream_t stream) {
  (void)in_sizes; (void)n_in; (void)out_size; (void)ws_size;
  const float* hidden    = (const float*)d_in[0];
  const float* weather   = (const float*)d_in[1];
  const float* in_proj_w = (const float*)d_in[2];
  const float* conv_w    = (const float*)d_in[3];
  const float* conv_b    = (const float*)d_in[4];
  const float* weather_w = (const float*)d_in[5];
  const float* weather_b = (const float*)d_in[6];
  const float* x_proj_w  = (const float*)d_in[7];
  const float* dt_proj_w = (const float*)d_in[8];
  const float* dt_proj_b = (const float*)d_in[9];
  const float* A_log     = (const float*)d_in[10];
  const float* D_param   = (const float*)d_in[11];
  const float* out_proj_w= (const float*)d_in[12];
  float* out = (float*)d_out;

  // carve workspace (256B aligned)
  char* ws = (char*)d_ws;
  size_t off = 0;
  auto carve = [&](size_t bytes) -> char* {
    char* p = ws + off;
    off = (off + bytes + 255) & ~(size_t)255;
    return p;
  };
  __bf16* hid_bf    = (__bf16*)carve((size_t)BL * D_ * 2);           // 8192 x 768
  __bf16* wea_bf    = (__bf16*)carve((size_t)BL * WW * 2);           // 8192 x 64
  __bf16* w_in_bf   = (__bf16*)carve((size_t)(2 * DI_) * D_ * 2);    // 3072 x 768
  __bf16* w_wea_bf  = (__bf16*)carve((size_t)WW * WW * 2);           // 64 x 64
  __bf16* w_xp_bf   = (__bf16*)carve((size_t)128 * (DI_ + WW) * 2);  // 80->128 x 1600
  __bf16* w_dt_bf   = (__bf16*)carve((size_t)DI_ * RRP * 2);         // 1536 x 48->64
  __bf16* w_out_bf  = (__bf16*)carve((size_t)D_ * DI_ * 2);          // 768 x 1536
  float*  xz_f      = (float*) carve((size_t)BL * (2 * DI_) * 4);
  float*  xconv_f   = (float*) carve((size_t)BL * DI_ * 4);
  float*  wp_f      = (float*) carve((size_t)BL * WW * 4);
  __bf16* concat_bf = (__bf16*)carve((size_t)BL * (DI_ + WW) * 2);   // 8192 x 1600
  float*  xdbl_f    = (float*) carve((size_t)BL * (RR + 2 * NST) * 4);
  __bf16* dtlow_bf  = (__bf16*)carve((size_t)BL * RRP * 2);          // 8192 x 48->64
  float*  delta_f   = (float*) carve((size_t)BL * DI_ * 4);
  __bf16* ygate_bf  = (__bf16*)carve((size_t)BL * DI_ * 2);

  auto pack = [&](const float* src, __bf16* dst, int rows, int cols,
                  int srcLd, int srcOff, int colsPad, int rowsPad) {
    const int total = rowsPad * colsPad;
    pack_pad_bf16<<<eblocks(total), 256, 0, stream>>>(src, dst, rows, cols,
                                                      srcLd, srcOff, colsPad, total);
  };
  // activations (no padding needed)
  pack(hidden,  hid_bf, BL, D_,  D_,  0, D_,  BL);
  pack(weather, wea_bf, BL, WW,  WW,  0, WW,  BL);
  // weights (zero-pad N to mult of 64, K to mult of 32)
  pack(in_proj_w,  w_in_bf,  2 * DI_, D_,        D_,        0, D_,        2 * DI_);
  pack(weather_w,  w_wea_bf, WW,      WW,        WW,        0, WW,        WW);
  pack(x_proj_w,   w_xp_bf,  RR+2*NST,DI_ + WW,  DI_ + WW,  0, DI_ + WW,  128);
  pack(dt_proj_w,  w_dt_bf,  DI_,     RR,        RR,        0, RRP,       DI_);
  pack(out_proj_w, w_out_bf, D_,      DI_,       DI_,       0, DI_,       D_);

  // 1) in_proj: xz(BL,3072) = hidden x in_proj_w^T
  wmma_gemm_bf16<<<gblocks(BL, 2 * DI_), 256, 0, stream>>>(
      hid_bf, w_in_bf, xz_f, nullptr, BL, 2 * DI_, 2 * DI_, D_, 0);

  // 2) causal depthwise conv + SiLU
  conv_silu_kernel<<<eblocks(BL * DI_), 256, 0, stream>>>(xz_f, conv_w, conv_b, xconv_f);

  // 3) weather projection (+bias)
  wmma_gemm_bf16<<<gblocks(BL, WW), 256, 0, stream>>>(
      wea_bf, w_wea_bf, wp_f, weather_b, BL, WW, WW, WW, 0);

  // 4) concat [x|wp] -> bf16, x_proj: x_dbl(BL,80)
  concat_pack_kernel<<<eblocks(BL * (DI_ + WW)), 256, 0, stream>>>(xconv_f, wp_f, concat_bf);
  wmma_gemm_bf16<<<gblocks(BL, 128), 256, 0, stream>>>(
      concat_bf, w_xp_bf, xdbl_f, nullptr, BL, RR + 2 * NST, 128, DI_ + WW, 0);

  // 5) dt_low (strided extract, K-pad to 64) -> dt GEMM + softplus
  pack(xdbl_f, dtlow_bf, BL, RR, RR + 2 * NST, 0, RRP, BL);
  wmma_gemm_bf16<<<gblocks(BL, DI_), 256, 0, stream>>>(
      dtlow_bf, w_dt_bf, delta_f, dt_proj_b, BL, DI_, DI_, RRP, 1);

  // 6) fused selective scan + D-skip + z-gate -> bf16 ygate
  scan_gate_kernel<<<(B_ * DI_ * NST) / 256, 256, 0, stream>>>(
      delta_f, xconv_f, xdbl_f, xz_f, A_log, D_param, ygate_bf);

  // 7) out_proj -> d_out (BL,768) f32
  wmma_gemm_bf16<<<gblocks(BL, D_), 256, 0, stream>>>(
      ygate_bf, w_out_bf, out, nullptr, BL, D_, D_, DI_, 0);
}
